// VisionTransformer_82025285419741
// MI455X (gfx1250) — compile-verified, matrix operand order checked by reading, not run on hardware
//
#include <hip/hip_runtime.h>
#include <hip/hip_bf16.h>

// ---------------------------------------------------------------------------
// ViT forward for MI455X (gfx1250). All GEMMs run on V_WMMA_F32_16X16X32_F16
// (wave32, fp16 operands, fp32 accumulate). ~2 TFLOP of matmul per call ->
// compute bound. 128x128 block tile, 8 waves of 64x32 each: 8 WMMA per
// K-step per wave at 0.75 fragment loads per WMMA. B tiles staged in LDS via
// gfx1250 async load-to-LDS (ASYNCcnt) -- builtin confirmed present, takes
// (int4 global*, int4 shared*, imm offset, imm cpol).
// ---------------------------------------------------------------------------

typedef __attribute__((ext_vector_type(16))) _Float16 v16h;
typedef __attribute__((ext_vector_type(8)))  _Float16 v8h;
typedef __attribute__((ext_vector_type(8)))  float    v8f;
typedef __attribute__((ext_vector_type(4)))  int      v4i;

#if defined(__HIP_DEVICE_COMPILE__) && \
    __has_builtin(__builtin_amdgcn_global_load_async_to_lds_b128) && \
    __has_builtin(__builtin_amdgcn_s_wait_asynccnt)
#define USE_ASYNC 1
typedef __attribute__((address_space(1))) v4i gv4i_t;
typedef __attribute__((address_space(3))) v4i lv4i_t;
#else
#define USE_ASYNC 0
#endif

#define TB    64      // batch
#define NTOK  197     // tokens
#define NPAT  196
#define DIM   768
#define NH    12
#define HD    64
#define NMLP  3072
#define NCLS  1000
#define NLAY  11
#define NPAD  224     // padded K for P@V (197 -> 224, zero filled)
#define CPAD  1024    // padded K for cls reinjection (1000 -> 1024)
#define ALPHA 0.6f
#define ATTSC 0.125f  // 64^-0.5
#define LNEPS 1e-6f

// ---------------- fragment load (row-major, ld multiple of 8) ---------------
__device__ inline v16h load_frag(const _Float16* base, int ld,
                                 int r0, int k0, int rmax) {
  int lane = threadIdx.x & 31;
  int r = r0 + (lane & 15);
  if (r >= rmax) r = rmax - 1;                 // clamp; stores are guarded
  int kk = k0 + ((lane >> 4) << 3);            // lanes 16-31 take K+8
  const _Float16* p = base + (long)r * ld + kk;
  v8h lo = *(const v8h*)(p);                   // K = kk .. kk+7
  v8h hi = *(const v8h*)(p + 16);              // K = kk+16 .. kk+23
  v16h out;
#pragma unroll
  for (int i = 0; i < 8; ++i) { out[i] = lo[i]; out[i + 8] = hi[i]; }
  return out;
}

// ---------------- generic WMMA GEMM: C[M,N] = A[M,K] * Bt[N,K]^T ------------
enum { EPI_F32 = 0, EPI_F16 = 1, EPI_F16_GELU = 2, EPI_RESID = 3,
       EPI_AXPY = 4, EPI_QKV = 5 };

template <int E>
__global__ __launch_bounds__(256) void gemm_wmma(
    const _Float16* __restrict__ A, long sAb, int lda,
    const _Float16* __restrict__ Bt, long sBb, int ldb,
    const float* __restrict__ bias, float scale,
    float* __restrict__ outF, _Float16* __restrict__ outH, long sOb, int ldc,
    int M, int N, int Kpad,
    _Float16* __restrict__ q16, _Float16* __restrict__ k16,
    _Float16* __restrict__ v16, _Float16* __restrict__ vT16) {
  const int batch = blockIdx.z;
  const _Float16* Ab = A + (long)batch * sAb;
  const _Float16* Bb = Bt + (long)batch * sBb;
  const int w  = threadIdx.x >> 5;
  const int wm = w >> 2, wn = w & 3;               // 2x4 waves, 64x32 each
  const int bn0 = blockIdx.x * 128;
  const int m0 = blockIdx.y * 128 + wm * 64;
  const int n0 = bn0 + wn * 32;
  const int lane = threadIdx.x & 31;

#if USE_ASYNC
  __shared__ _Float16 ldsB[128 * 32];              // 8 KB B tile (N x Kstep)
#endif

  v8f acc[4][2];
  {
    v8f z = {0.f, 0.f, 0.f, 0.f, 0.f, 0.f, 0.f, 0.f};
#pragma unroll
    for (int mi = 0; mi < 4; ++mi) { acc[mi][0] = z; acc[mi][1] = z; }
  }

  for (int k0 = 0; k0 < Kpad; k0 += 32) {
#if USE_ASYNC
    __syncthreads();                               // prior consumers done
    {
      int t = threadIdx.x;
#pragma unroll
      for (int c = 0; c < 2; ++c) {
        int chunk = t * 2 + c;                     // 0..511 : 128 rows x 4
        int rowI = chunk >> 2, sub = chunk & 3;
        int n = bn0 + rowI; if (n >= N) n = N - 1;
        const _Float16* src = Bb + (long)n * ldb + k0 + sub * 8;
        _Float16* dst = &ldsB[rowI * 32 + sub * 8];
        __builtin_amdgcn_global_load_async_to_lds_b128(
            (gv4i_t*)src, (lv4i_t*)dst, 0, 0);
      }
      __builtin_amdgcn_s_wait_asynccnt(0);
    }
    __syncthreads();                               // tile visible to all waves
    v16h b0 = load_frag(ldsB, 32, wn * 32,      0, 1 << 30);
    v16h b1 = load_frag(ldsB, 32, wn * 32 + 16, 0, 1 << 30);
#else
    v16h b0 = load_frag(Bb, ldb, n0,      k0, N);
    v16h b1 = load_frag(Bb, ldb, n0 + 16, k0, N);
#endif
    if (k0 + 32 < Kpad) {                          // prefetch next A K-slice
      int pr = m0 + (lane & 15); if (pr >= M) pr = M - 1;
      __builtin_prefetch(Ab + (long)pr * lda + k0 + 32, 0, 3);
    }
#pragma unroll
    for (int mi = 0; mi < 4; ++mi) {
      v16h a = load_frag(Ab, lda, m0 + mi * 16, k0, M);
      acc[mi][0] = __builtin_amdgcn_wmma_f32_16x16x32_f16(false, a, false, b0,
                                                          (short)0, acc[mi][0], false, false);
      acc[mi][1] = __builtin_amdgcn_wmma_f32_16x16x32_f16(false, a, false, b1,
                                                          (short)0, acc[mi][1], false, false);
    }
  }

  const int rAdd = (lane >> 4) << 3;               // C layout: lanes>=16 -> M+8
#pragma unroll
  for (int mi = 0; mi < 4; ++mi) {
#pragma unroll
    for (int ni = 0; ni < 2; ++ni) {
      int col = n0 + ni * 16 + (lane & 15);
      bool cok = (col < N);
      float bb = (bias && cok) ? bias[col] : 0.f;
#pragma unroll
      for (int r = 0; r < 8; ++r) {
        int row = m0 + mi * 16 + rAdd + r;
        if (row >= M || !cok) continue;
        float val = acc[mi][ni][r];
        if constexpr (E == EPI_QKV) {
          int b_  = row / NTOK, n_ = row % NTOK;
          int tri = col / DIM;  int rem = col - tri * DIM;
          int hh  = rem >> 6;   int hd = rem & 63;
          _Float16 hv = (_Float16)(val + bb);
          long idx = (((long)(b_ * NH + hh) * NTOK + n_) << 6) + hd;
          if (tri == 0)      q16[idx] = hv;
          else if (tri == 1) k16[idx] = hv;
          else { v16[idx] = hv;
                 vT16[((long)(b_ * NH + hh) * HD + hd) * NPAD + n_] = hv; }
        } else {
          long idx = (long)batch * sOb + (long)row * ldc + col;
          if constexpr (E == EPI_F32)       outF[idx] = val * scale + bb;
          else if constexpr (E == EPI_F16)  outH[idx] = (_Float16)(val + bb);
          else if constexpr (E == EPI_F16_GELU) {
            float t = val + bb;
            outH[idx] = (_Float16)(0.5f * t * (1.f + erff(t * 0.70710678118f)));
          }
          else if constexpr (E == EPI_RESID) outF[idx] += val + bb;
          else if constexpr (E == EPI_AXPY)  outF[idx] += scale * (val + bb);
        }
      }
    }
  }
}

// ---------------- LayerNorm: fp32 in -> fp16 out (one row per block) --------
__global__ __launch_bounds__(256) void ln_kernel(
    const float* __restrict__ in, long inRow,
    const float* __restrict__ g, const float* __restrict__ be,
    _Float16* __restrict__ out, long outRow, int cols) {
  long row = blockIdx.x;
  const float* p = in + row * inRow;
  __shared__ float s1[256], s2[256];
  float a = 0.f, b = 0.f;
  for (int c = threadIdx.x; c < cols; c += 256) { float v = p[c]; a += v; b += v * v; }
  s1[threadIdx.x] = a; s2[threadIdx.x] = b; __syncthreads();
  for (int st = 128; st > 0; st >>= 1) {
    if (threadIdx.x < (unsigned)st) { s1[threadIdx.x] += s1[threadIdx.x + st];
                                      s2[threadIdx.x] += s2[threadIdx.x + st]; }
    __syncthreads();
  }
  float mean = s1[0] / cols;
  float var  = s2[0] / cols - mean * mean;
  float rstd = rsqrtf(var + LNEPS);
  for (int c = threadIdx.x; c < cols; c += 256)
    out[row * outRow + c] = (_Float16)((p[c] - mean) * rstd * g[c] + be[c]);
}

// ---------------- softmax over 197, fp32 -> fp16 row padded to 224 ----------
__global__ __launch_bounds__(256) void softmax_kernel(
    const float* __restrict__ S, _Float16* __restrict__ P) {
  long row = blockIdx.x;
  const float* p = S + row * NTOK;
  _Float16* o = P + row * NPAD;
  int t = threadIdx.x;
  __shared__ float red[256];
  float v = (t < NTOK) ? p[t] : -3.0e38f;
  red[t] = v; __syncthreads();
  for (int st = 128; st > 0; st >>= 1) {
    if (t < st) red[t] = fmaxf(red[t], red[t + st]);
    __syncthreads();
  }
  float mx = red[0]; __syncthreads();
  float e = (t < NTOK) ? expf(v - mx) : 0.f;
  red[t] = e; __syncthreads();
  for (int st = 128; st > 0; st >>= 1) {
    if (t < st) red[t] += red[t + st];
    __syncthreads();
  }
  float inv = 1.f / red[0];
  if (t < NPAD) o[t] = (_Float16)((t < NTOK) ? e * inv : 0.f);
}

// ---------------- small elementwise kernels ---------------------------------
__global__ void tconv_kernel(const float* __restrict__ w, _Float16* __restrict__ wt,
                             int K, int N, int ldt, long total) {
  long i = (long)blockIdx.x * 256 + threadIdx.x;
  if (i >= total) return;
  int k = (int)(i / N), n = (int)(i % N);
  wt[(long)n * ldt + k] = (_Float16)w[i];
}

__global__ void conv_kernel(const float* __restrict__ w, _Float16* __restrict__ o,
                            long total) {
  long i = (long)blockIdx.x * 256 + threadIdx.x;
  if (i < total) o[i] = (_Float16)w[i];
}

__global__ void im2col_kernel(const float* __restrict__ x, _Float16* __restrict__ A,
                              long total) {
  long i = (long)blockIdx.x * 256 + threadIdx.x;
  if (i >= total) return;
  int k = (int)(i % DIM);            // k = c*256 + p*16 + q
  long rowi = i / DIM;               // row = (b, hy, wx)
  int c = k >> 8, rem = k & 255, p = rem >> 4, q = rem & 15;
  int b = (int)(rowi / NPAT), t = (int)(rowi % NPAT);
  int hy = t / 14, wx = t % 14;
  long src = (((long)(b * 3 + c) * 224 + hy * 16 + p) * 224) + wx * 16 + q;
  A[i] = (_Float16)x[src];
}

__global__ void assemble_kernel(float* __restrict__ x, const float* __restrict__ tok,
                                const float* __restrict__ clsTok,
                                const float* __restrict__ pos,
                                const float* __restrict__ patch_b, long total) {
  long i = (long)blockIdx.x * 256 + threadIdx.x;
  if (i >= total) return;
  int d = (int)(i % DIM);
  long rowi = i / DIM;
  int n = (int)(rowi % NTOK), b = (int)(rowi / NTOK);
  float v;
  if (n == 0) v = clsTok[d] + pos[d];
  else        v = tok[((long)b * NPAT + (n - 1)) * DIM + d] + patch_b[d] + pos[(long)n * DIM + d];
  x[i] = v;
}

__global__ void vdiff_kernel(const _Float16* __restrict__ v0,
                             const float* __restrict__ gv,
                             _Float16* __restrict__ d16, long total) {
  long i = (long)blockIdx.x * 256 + threadIdx.x;
  if (i < total) d16[i] = (_Float16)((float)v0[i] - gv[i]);
}

__global__ void merge_heads_kernel(const float* __restrict__ O,
                                   _Float16* __restrict__ om, long total) {
  long i = (long)blockIdx.x * 256 + threadIdx.x;
  if (i >= total) return;
  int hd = (int)(i & 63);
  long r = i >> 6;
  int n = (int)(r % NTOK); r /= NTOK;
  int h = (int)(r % NH);
  int b = (int)(r / NH);
  om[((long)(b * NTOK + n) * DIM) + h * HD + hd] = (_Float16)O[i];
}

// ---------------- host side -------------------------------------------------
template <int E>
static void launch_gemm(hipStream_t st, const void* A, long sA, int lda,
                        const void* Bt, long sB, int ldb,
                        const float* bias, float scale,
                        void* outF, void* outH, long sO, int ldc,
                        int M, int N, int Kpad, int batches,
                        void* q = nullptr, void* k = nullptr,
                        void* v = nullptr, void* vT = nullptr) {
  dim3 g((unsigned)((N + 127) / 128), (unsigned)((M + 127) / 128), (unsigned)batches);
  gemm_wmma<E><<<g, dim3(256), 0, st>>>(
      (const _Float16*)A, sA, lda, (const _Float16*)Bt, sB, ldb, bias, scale,
      (float*)outF, (_Float16*)outH, sO, ldc, M, N, Kpad,
      (_Float16*)q, (_Float16*)k, (_Float16*)v, (_Float16*)vT);
}

static inline dim3 grid1d(long total) { return dim3((unsigned)((total + 255) / 256)); }

extern "C" void kernel_launch(void* const* d_in, const int* in_sizes, int n_in,
                              void* d_out, int out_size, void* d_ws, size_t ws_size,
                              hipStream_t stream) {
  (void)in_sizes; (void)n_in; (void)out_size; (void)ws_size;
  // ---- inputs (setup_inputs order) ----
  const float* xin      = (const float*)d_in[0];
  const float* patch_w  = (const float*)d_in[1];
  const float* patch_b  = (const float*)d_in[2];
  const float* cls_tok  = (const float*)d_in[3];
  const float* pos_emb  = (const float*)d_in[4];
  const float* qkv_w    = (const float*)d_in[5];
  const float* qkv_b    = (const float*)d_in[6];
  const float* proj_w   = (const float*)d_in[7];
  const float* proj_b   = (const float*)d_in[8];
  const float* gconv_w  = (const float*)d_in[9];
  const float* gconv_b  = (const float*)d_in[10];
  const float* n1_g     = (const float*)d_in[11];
  const float* n1_b     = (const float*)d_in[12];
  const float* n2_g     = (const float*)d_in[13];
  const float* n2_b     = (const float*)d_in[14];
  const float* fc1_w    = (const float*)d_in[15];
  const float* fc1_b    = (const float*)d_in[16];
  const float* fc2_w    = (const float*)d_in[17];
  const float* fc2_b    = (const float*)d_in[18];
  const float* norm_g   = (const float*)d_in[19];
  const float* norm_b   = (const float*)d_in[20];
  const float* head_w   = (const float*)d_in[21];
  const float* head_b   = (const float*)d_in[22];
  const float* rev_w    = (const float*)d_in[23];
  const float* rev_b    = (const float*)d_in[24];
  float* out            = (float*)d_out;

  // ---- workspace carve (~625 MB total) ----
  const long TOKENS = (long)TB * NTOK;           // 12608
  const long TOKEL  = TOKENS * DIM;              // 9,682,944
  const long BH     = (long)TB * NH;             // 768
  const long ATTEL  = BH * NTOK * NTOK;          // 29,805,312
  const long PPAD   = BH * NTOK * NPAD;          // 33,890,304
  const long VEL    = BH * NTOK * HD;            // 9,682,944
  const long VTEL   = BH * HD * NPAD;            // 11,010,048

  char* wsb = (char*)d_ws; size_t off = 0;
  auto carve = [&](long elems, int esz) -> void* {
    off = (off + 255) & ~(size_t)255;
    void* p = wsb + off; off += (size_t)elems * esz; return p;
  };
  // fp16 weights (transposed for Bt layout)
  _Float16* qkv_wt  = (_Float16*)carve((long)NLAY * 3 * DIM * DIM, 2);
  _Float16* proj_wt = (_Float16*)carve((long)NLAY * DIM * DIM, 2);
  _Float16* fc1_wt  = (_Float16*)carve((long)NLAY * NMLP * DIM, 2);
  _Float16* fc2_wt  = (_Float16*)carve((long)NLAY * DIM * NMLP, 2);
  _Float16* gconv16 = (_Float16*)carve((long)NLAY * HD * HD, 2);
  _Float16* head_wt = (_Float16*)carve((long)NCLS * DIM, 2);
  _Float16* rev_wt  = (_Float16*)carve((long)DIM * CPAD, 2);
  _Float16* patch16 = (_Float16*)carve((long)DIM * DIM, 2);
  // activations
  float*    x    = (float*)carve(TOKEL, 4);
  _Float16* h16  = (_Float16*)carve(TOKEL, 2);   // also reused as im2col A
  _Float16* q16  = (_Float16*)carve(VEL, 2);
  _Float16* k16  = (_Float16*)carve(VEL, 2);
  _Float16* v16  = (_Float16*)carve(VEL, 2);
  _Float16* v0_16= (_Float16*)carve(VEL, 2);
  _Float16* vT16 = (_Float16*)carve(VTEL, 2);
  float*    Sbuf = (float*)carve(ATTEL, 4);      // scores; reused: tok, gv
  _Float16* P16  = (_Float16*)carve(PPAD, 2);    // softmax probs; reused: d16
  float*    Obuf = (float*)carve(VEL, 4);        // attention output [B,H,N,HD]
  _Float16* om16 = (_Float16*)carve(TOKEL, 2);   // merged heads fp16
  _Float16* mid16= (_Float16*)carve(TOKENS * (long)NMLP, 2);
  _Float16* cls16= (_Float16*)carve((long)TB * DIM, 2);
  _Float16* t16  = (_Float16*)carve((long)TB * CPAD, 2);
  float*    tokF = Sbuf;                         // patch-embed output (reuse)
  float*    gvF  = Sbuf;                         // gconv temp (reuse)
  _Float16* d16  = P16;                          // v0-gv temp (reuse)

  // ---- zero the zero-padded regions (pads are never written elsewhere) ----
  (void)hipMemsetAsync(vT16, 0, (size_t)VTEL * 2, stream);
  (void)hipMemsetAsync(t16, 0, (size_t)TB * CPAD * 2, stream);
  (void)hipMemsetAsync(rev_wt, 0, (size_t)DIM * CPAD * 2, stream);

  // ---- weight conversion (fp32 -> fp16, transposed to [N,K]) ----
  for (int i = 0; i < NLAY; ++i) {
    long s;
    s = (long)DIM * 3 * DIM;
    tconv_kernel<<<grid1d(s), 256, 0, stream>>>(qkv_w + i * s, qkv_wt + (long)i * 3 * DIM * DIM, DIM, 3 * DIM, DIM, s);
    s = (long)DIM * DIM;
    tconv_kernel<<<grid1d(s), 256, 0, stream>>>(proj_w + i * s, proj_wt + i * s, DIM, DIM, DIM, s);
    s = (long)DIM * NMLP;
    tconv_kernel<<<grid1d(s), 256, 0, stream>>>(fc1_w + i * s, fc1_wt + i * s, DIM, NMLP, DIM, s);
    tconv_kernel<<<grid1d(s), 256, 0, stream>>>(fc2_w + i * s, fc2_wt + i * s, NMLP, DIM, NMLP, s);
    s = (long)HD * HD;     // gconv_w[o,i] is already Bt layout: plain convert
    conv_kernel<<<grid1d(s), 256, 0, stream>>>(gconv_w + i * s, gconv16 + i * s, s);
  }
  { long s = (long)DIM * NCLS;
    tconv_kernel<<<grid1d(s), 256, 0, stream>>>(head_w, head_wt, DIM, NCLS, DIM, s); }
  { long s = (long)NCLS * DIM;
    tconv_kernel<<<grid1d(s), 256, 0, stream>>>(rev_w, rev_wt, NCLS, DIM, CPAD, s); }
  { long s = (long)DIM * DIM; // patch_w[d, cpq] is already Bt layout
    conv_kernel<<<grid1d(s), 256, 0, stream>>>(patch_w, patch16, s); }

  // ---- patch embed ----
  { long s = (long)TB * NPAT * DIM;
    im2col_kernel<<<grid1d(s), 256, 0, stream>>>(xin, h16, s);
    launch_gemm<EPI_F32>(stream, h16, 0, DIM, patch16, 0, DIM, nullptr, 1.f,
                         tokF, nullptr, 0, DIM, (int)(TB * NPAT), DIM, DIM, 1);
    assemble_kernel<<<grid1d(TOKEL), 256, 0, stream>>>(x, tokF, cls_tok, pos_emb, patch_b, TOKEL);
  }

  // ---- transformer blocks ----
  for (int i = 0; i < NLAY; ++i) {
    // LN1 -> h16
    ln_kernel<<<dim3((unsigned)TOKENS), 256, 0, stream>>>(x, DIM, n1_g + i * DIM, n1_b + i * DIM, h16, DIM, DIM);
    // QKV GEMM with scatter epilogue
    launch_gemm<EPI_QKV>(stream, h16, 0, DIM, qkv_wt + (long)i * 3 * DIM * DIM, 0, DIM,
                         qkv_b + i * 3 * DIM, 1.f, nullptr, nullptr, 0, 0,
                         (int)TOKENS, 3 * DIM, DIM, 1, q16, k16, v16, vT16);
    if (i == 0)
      (void)hipMemcpyAsync(v0_16, v16, (size_t)VEL * 2, hipMemcpyDeviceToDevice, stream);
    // scores: S = scale * q @ k^T, batched over B*H
    launch_gemm<EPI_F32>(stream, q16, (long)NTOK * HD, HD, k16, (long)NTOK * HD, HD,
                         nullptr, ATTSC, Sbuf, nullptr, (long)NTOK * NTOK, NTOK,
                         NTOK, NTOK, HD, (int)BH);
    // softmax + fp16 pack (rows padded to 224 with zeros)
    softmax_kernel<<<dim3((unsigned)(BH * NTOK)), 256, 0, stream>>>(Sbuf, P16);
    // O = P @ V (Bt = v transposed, zero-padded K)
    launch_gemm<EPI_F32>(stream, P16, (long)NTOK * NPAD, NPAD, vT16, (long)HD * NPAD, NPAD,
                         nullptr, 1.f, Obuf, nullptr, (long)NTOK * HD, HD,
                         NTOK, HD, NPAD, (int)BH);
    if (i > 0) {
      // gv = v @ W^T + b     (M = B*H*N rows of 64)
      launch_gemm<EPI_F32>(stream, v16, 0, HD, gconv16 + (long)i * HD * HD, 0, HD,
                           gconv_b + i * HD, 1.f, gvF, nullptr, 0, HD,
                           (int)(BH * NTOK), HD, HD, 1);
      vdiff_kernel<<<grid1d(VEL), 256, 0, stream>>>(v0_16, gvF, d16, VEL);
      // O += alpha * ((v0-gv) @ W^T + b)
      launch_gemm<EPI_AXPY>(stream, d16, 0, HD, gconv16 + (long)i * HD * HD, 0, HD,
                            gconv_b + i * HD, ALPHA, Obuf, nullptr, 0, HD,
                            (int)(BH * NTOK), HD, HD, 1);
    }
    // merge heads -> fp16 [B*N, D]
    merge_heads_kernel<<<grid1d(VEL), 256, 0, stream>>>(Obuf, om16, VEL);
    // proj + residual into x
    launch_gemm<EPI_RESID>(stream, om16, 0, DIM, proj_wt + (long)i * DIM * DIM, 0, DIM,
                           proj_b + i * DIM, 1.f, x, nullptr, 0, DIM,
                           (int)TOKENS, DIM, DIM, 1);
    // LN2 -> h16
    ln_kernel<<<dim3((unsigned)TOKENS), 256, 0, stream>>>(x, DIM, n2_g + i * DIM, n2_b + i * DIM, h16, DIM, DIM);
    // fc1 + exact GELU -> fp16 mid
    launch_gemm<EPI_F16_GELU>(stream, h16, 0, DIM, fc1_wt + (long)i * NMLP * DIM, 0, DIM,
                              fc1_b + i * NMLP, 1.f, nullptr, mid16, 0, NMLP,
                              (int)TOKENS, NMLP, DIM, 1);
    // fc2 + residual into x
    launch_gemm<EPI_RESID>(stream, mid16, 0, NMLP, fc2_wt + (long)i * DIM * NMLP, 0, NMLP,
                           fc2_b + i * DIM, 1.f, x, nullptr, 0, DIM,
                           (int)TOKENS, DIM, NMLP, 1);
    if (i >= 1) {
      // mid-layer cls prediction reinjected into cls token
      ln_kernel<<<dim3((unsigned)TB), 256, 0, stream>>>(x, (long)NTOK * DIM, norm_g, norm_b, cls16, DIM, DIM);
      launch_gemm<EPI_F16>(stream, cls16, 0, DIM, head_wt, 0, DIM,
                           head_b, 1.f, nullptr, t16, 0, CPAD, TB, NCLS, DIM, 1);
      launch_gemm<EPI_F32>(stream, t16, 0, CPAD, rev_wt, 0, CPAD,
                           rev_b, 1.f, x, nullptr, 0, (int)((long)NTOK * DIM),
                           TB, DIM, CPAD, 1);
    }
  }

  // ---- final head ----
  ln_kernel<<<dim3((unsigned)TB), 256, 0, stream>>>(x, (long)NTOK * DIM, norm_g, norm_b, cls16, DIM, DIM);
  launch_gemm<EPI_F32>(stream, cls16, 0, DIM, head_wt, 0, DIM,
                       head_b, 1.f, out, nullptr, 0, NCLS, TB, NCLS, DIM, 1);
}
